// CGCNNPyG_74637941670355
// MI455X (gfx1250) — matrix operand order repaired and checked by reading, not compiled, not run on hardware
//
#include <hip/hip_runtime.h>

// ---------------------------------------------------------------------------
// Types for CDNA5 WMMA
// ---------------------------------------------------------------------------
typedef _Float16 f16;
typedef __attribute__((ext_vector_type(16))) _Float16 v16h;
typedef __attribute__((ext_vector_type(8)))  _Float16 v8h;
typedef __attribute__((ext_vector_type(4)))  _Float16 v4h;
typedef __attribute__((ext_vector_type(8)))  float    v8f;
typedef __attribute__((ext_vector_type(4)))  float    v4f;

__device__ __forceinline__ v8f wmma16(v16h a, v16h b, v8f c) {
    // D = A(16x32 f16) * B(32x16 f16) + C(16x16 f32)
    return __builtin_amdgcn_wmma_f32_16x16x32_f16(
        /*neg_a=*/false, a, /*neg_b=*/false, b,
        /*c_mod=*/(short)0, c, /*reuse_a=*/false, /*reuse_b=*/false);
}

// Softplus straight on the hardware transcendentals (v_exp_f32 = 2^x,
// v_log_f32 = log2):  sp(x) = log2(1 + 2^(x*log2e)) * ln2, branchless guard.
__device__ __forceinline__ float softplus_f(float x) {
    const float LOG2E = 1.44269504088896340736f;
    const float LN2   = 0.69314718055994530942f;
    float t = __builtin_amdgcn_logf(1.0f + __builtin_amdgcn_exp2f(x * LOG2E)) * LN2;
    return (x > 20.0f) ? x : t;
}

// A fragment: 16x32 f16 tile, row-major in LDS with row stride `stride` (f16 units).
// lane L: m = L%16, kb = (L/16)*8. a[0..7] = K[ks*32+kb .. +7], a[8..15] = K[ks*32+16+kb .. +7]
__device__ __forceinline__ v16h afrag(const f16* __restrict__ buf, int stride, int ks) {
    int lane = threadIdx.x & 31;
    int m    = lane & 15;
    int kb   = (lane >> 4) * 8 + ks * 32;
    const f16* p = buf + m * stride + kb;
    v16h a;
#pragma unroll
    for (int i = 0; i < 8; ++i) a[i] = p[i];
#pragma unroll
    for (int i = 0; i < 8; ++i) a[8 + i] = p[16 + i];
    return a;
}

// B fragment from transposed weights Bt[n][k] (row stride K f16).
// lane L: n = nbase + L%16, kb = ks*32 + (L/16)*8, b[0..7] = Bt[n][kb .. +7], b[8..15] = Bt[n][kb+16 .. +7]
__device__ __forceinline__ v16h bfrag(const f16* __restrict__ bt, int K, int ks, int nbase) {
    int lane = threadIdx.x & 31;
    int n    = nbase + (lane & 15);
    int kb   = ks * 32 + ((lane >> 4) << 3);
    const f16* p = bt + n * K + kb;
    v16h b;
#pragma unroll
    for (int i = 0; i < 8; ++i) b[i] = p[i];
#pragma unroll
    for (int i = 0; i < 8; ++i) b[8 + i] = p[16 + i];
    return b;
}

// Packed transposed-weights blob per layer (f16 units):
//   euW1t [64][192] | euW2t [64][64] | nuW1t [64][128] | nuW2t [64][64]
#define W_EU1   (64 * 192)
#define W_EU2   (64 * 64)
#define W_NU1   (64 * 128)
#define W_NU2   (64 * 64)
#define W_TOTAL (W_EU1 + W_EU2 + W_NU1 + W_NU2)   // 28672 f16 per layer

// ---------------------------------------------------------------------------
// One-time weight pack: convert+transpose all 3 layers' MLP weights to f16.
// ---------------------------------------------------------------------------
__global__ void pack_weights_kernel(const float* __restrict__ euW1,
                                    const float* __restrict__ euW2,
                                    const float* __restrict__ nuW1,
                                    const float* __restrict__ nuW2,
                                    f16* __restrict__ wpack, int layers)
{
    int idx = blockIdx.x * blockDim.x + threadIdx.x;
    int total = layers * W_TOTAL;
    if (idx >= total) return;
    int layer = idx / W_TOTAL;
    int off   = idx - layer * W_TOTAL;
    float v;
    if (off < W_EU1) {
        int n = off / 192, k = off % 192;
        v = euW1[layer * 192 * 64 + k * 64 + n];
    } else if (off < W_EU1 + W_EU2) {
        int o = off - W_EU1; int n = o / 64, k = o % 64;
        v = euW2[layer * 64 * 64 + k * 64 + n];
    } else if (off < W_EU1 + W_EU2 + W_NU1) {
        int o = off - (W_EU1 + W_EU2); int n = o / 128, k = o % 128;
        v = nuW1[layer * 128 * 64 + k * 64 + n];
    } else {
        int o = off - (W_EU1 + W_EU2 + W_NU1); int n = o / 64, k = o % 64;
        v = nuW2[layer * 64 * 64 + k * 64 + n];
    }
    wpack[idx] = (f16)v;
}

// ---------------------------------------------------------------------------
// Fused per-layer edge kernel: edge-update MLP + message MLP + scatter-add.
// One wave handles 16 edges; WPB waves per block.
// ---------------------------------------------------------------------------
#define WPB 8

__global__ __launch_bounds__(WPB * 32)
void edge_layer_kernel(const float* __restrict__ h,
                       float* e,                       // read + write (per-row, in place)
                       const int* __restrict__ rowI,
                       const int* __restrict__ colI,
                       const f16* __restrict__ wpack,  // this layer's packed weights
                       const float* __restrict__ eub1, const float* __restrict__ eub2,
                       const float* __restrict__ nub1, const float* __restrict__ nub2,
                       float* __restrict__ hnew,
                       int Etot)
{
    extern __shared__ char smem[];
    f16* wsm    = (f16*)smem;                   // W_TOTAL packed weights
    f16* euW1t  = wsm;                          // [64][192]
    f16* euW2t  = euW1t + W_EU1;                // [64][64]
    f16* nuW1t  = euW2t + W_EU2;                // [64][128]
    f16* nuW2t  = nuW1t + W_NU1;                // [64][64]
    f16* einAll = wsm + W_TOTAL;                // WPB * [16][192]
    f16* hidAll = einAll + WPB * 16 * 192;      // WPB * [16][64]
    int* rcAll  = (int*)(hidAll + WPB * 16 * 64); // WPB * [32] (row[0..15], col[0..15])

    const int tid = threadIdx.x;
    const int nth = blockDim.x;

    // Block-copy packed f16 weights global -> LDS (16B vector chunks, coalesced).
    {
        const v8h* src = (const v8h*)wpack;
        v8h* dst = (v8h*)wsm;
        for (int i = tid; i < W_TOTAL / 8; i += nth) dst[i] = src[i];
    }
    __syncthreads();

    const int wave = tid >> 5;
    const int lane = tid & 31;
    f16* ein = einAll + wave * 16 * 192;
    f16* hid = hidAll + wave * 16 * 64;
    int* rcR = rcAll + wave * 32;
    int* rcC = rcR + 16;

    const long base = ((long)blockIdx.x * WPB + wave) * 16;
    if (base >= Etot) return;

    // Stage row/col indices for this tile (clamped for tail).
    if (lane < 16) {
        long eIdx = base + lane; if (eIdx >= Etot) eIdx = Etot - 1;
        rcR[lane] = rowI[eIdx];
        rcC[lane] = colI[eIdx];
    }

    // Gather ein = [ h[row] | h[col] | e ] as f16, tile [16][192].
    // Vectorized: float4 global loads -> packed 4xf16 LDS stores (16 chunks/region).
    for (int idx = lane; idx < 16 * 48; idx += 32) {
        int j = idx / 48, q = idx - j * 48;
        int region = q >> 4;           // 0: h[row], 1: h[col], 2: e
        int co = (q & 15) << 2;        // float offset within 64-wide row
        const float* src;
        if (region == 0) {
            src = h + (long)rcR[j] * 64 + co;
        } else if (region == 1) {
            src = h + (long)rcC[j] * 64 + co;
        } else {
            long eIdx = base + j; if (eIdx >= Etot) eIdx = Etot - 1;
            src = e + eIdx * 64 + co;
        }
        v4f v = *(const v4f*)src;
        v4h hv;
#pragma unroll
        for (int i = 0; i < 4; ++i) hv[i] = (f16)v[i];
        *(v4h*)(ein + j * 192 + q * 4) = hv;
    }

    const int n0 = lane & 15;           // column within 16-wide N tile
    const int mh = (lane >> 4) * 8;     // row base for accumulator rows

    // ---- GEMM1: hidden = sp(ein[16x192] @ euW1 + b1) ----
    v16h aE[6];
#pragma unroll
    for (int ks = 0; ks < 6; ++ks) aE[ks] = afrag(ein, 192, ks);
#pragma unroll
    for (int t = 0; t < 4; ++t) {
        float bias = eub1[t * 16 + n0];
        v8f acc;
#pragma unroll
        for (int r = 0; r < 8; ++r) acc[r] = bias;
#pragma unroll
        for (int ks = 0; ks < 6; ++ks) acc = wmma16(aE[ks], bfrag(euW1t, 192, ks, t * 16), acc);
#pragma unroll
        for (int r = 0; r < 8; ++r)
            hid[(mh + r) * 64 + t * 16 + n0] = (f16)softplus_f(acc[r]);
    }

    // ---- GEMM2: e_new = hid[16x64] @ euW2 + b2 ----
    {
        v16h aH0 = afrag(hid, 64, 0), aH1 = afrag(hid, 64, 1);
#pragma unroll
        for (int t = 0; t < 4; ++t) {
            float bias = eub2[t * 16 + n0];
            v8f acc;
#pragma unroll
            for (int r = 0; r < 8; ++r) acc[r] = bias;
            acc = wmma16(aH0, bfrag(euW2t, 64, 0, t * 16), acc);
            acc = wmma16(aH1, bfrag(euW2t, 64, 1, t * 16), acc);
#pragma unroll
            for (int r = 0; r < 8; ++r) {
                int m = mh + r;
                long eIdx = base + m;
                float v = acc[r];
                if (eIdx < Etot) e[eIdx * 64 + t * 16 + n0] = v;    // persist e for next layer
                ein[m * 192 + 64 + t * 16 + n0] = (f16)v;           // splice into message input
            }
        }
    }

    // ---- GEMM3: mhid = sp(cat(h[row], e_new)[16x128] @ nuW1 + b1) ----
    v16h aM[4];
#pragma unroll
    for (int ks = 0; ks < 4; ++ks) aM[ks] = afrag(ein, 192, ks);   // K window [0,128)
#pragma unroll
    for (int t = 0; t < 4; ++t) {
        float bias = nub1[t * 16 + n0];
        v8f acc;
#pragma unroll
        for (int r = 0; r < 8; ++r) acc[r] = bias;
#pragma unroll
        for (int ks = 0; ks < 4; ++ks) acc = wmma16(aM[ks], bfrag(nuW1t, 128, ks, t * 16), acc);
#pragma unroll
        for (int r = 0; r < 8; ++r)
            hid[(mh + r) * 64 + t * 16 + n0] = (f16)softplus_f(acc[r]);
    }

    // ---- GEMM4: msg = mhid[16x64] @ nuW2 + b2, scatter-add to hnew[col] ----
    {
        v16h aH0 = afrag(hid, 64, 0), aH1 = afrag(hid, 64, 1);
#pragma unroll
        for (int t = 0; t < 4; ++t) {
            float bias = nub2[t * 16 + n0];
            v8f acc;
#pragma unroll
            for (int r = 0; r < 8; ++r) acc[r] = bias;
            acc = wmma16(aH0, bfrag(nuW2t, 64, 0, t * 16), acc);
            acc = wmma16(aH1, bfrag(nuW2t, 64, 1, t * 16), acc);
#pragma unroll
            for (int r = 0; r < 8; ++r) {
                int m = mh + r;
                if (base + m < Etot) {
                    int cdst = rcC[m];
                    atomicAdd(&hnew[(long)cdst * 64 + t * 16 + n0], acc[r]);
                }
            }
        }
    }
}

// ---------------------------------------------------------------------------
// Embedding kernels (odd K; memory-bound -> plain VALU)
// ---------------------------------------------------------------------------
__global__ void atom_embed_kernel(const float* __restrict__ x,
                                  const float* __restrict__ W,
                                  const float* __restrict__ b,
                                  float* __restrict__ h, int Nn)
{
    long i = (long)blockIdx.x * blockDim.x + threadIdx.x;
    if (i >= (long)Nn * 64) return;
    int r = (int)(i >> 6), c = (int)(i & 63);
    float acc = b[c];
    const float* xr = x + (long)r * 92;
#pragma unroll 4
    for (int k = 0; k < 92; ++k) acc += xr[k] * W[k * 64 + c];
    h[i] = acc;
}

__global__ void bond_embed_kernel(const float* __restrict__ ea,
                                  const float* __restrict__ W,
                                  const float* __restrict__ b,
                                  float* __restrict__ e, long Etot)
{
    long i = (long)blockIdx.x * blockDim.x + threadIdx.x;
    if (i >= Etot * 64) return;
    long r = i >> 6; int c = (int)(i & 63);
    float acc = b[c];
    const float* er = ea + r * 41;
#pragma unroll
    for (int k = 0; k < 41; ++k) acc += er[k] * W[k * 64 + c];
    e[i] = acc;
}

// ---------------------------------------------------------------------------
// BatchNorm (training-mode batch stats) + softplus + residual
// ---------------------------------------------------------------------------
__global__ void bn_reduce_kernel(const float* __restrict__ hn,
                                 float* __restrict__ psum, float* __restrict__ psq, int Nn)
{
    __shared__ float ls[256], lq[256];
    int tid = threadIdx.x;
    int c = tid & 63;
    float s = 0.f, q = 0.f;
    long total = (long)Nn * 64;
    for (long i = (long)blockIdx.x * blockDim.x + tid; i < total; i += (long)gridDim.x * blockDim.x) {
        float v = hn[i];
        s += v; q += v * v;
    }
    ls[tid] = s; lq[tid] = q;
    __syncthreads();
    if (tid < 64) {
#pragma unroll
        for (int o = 64; o < 256; o += 64) { s += ls[tid + o]; q += lq[tid + o]; }
        atomicAdd(&psum[c], s);
        atomicAdd(&psq[c], q);
    }
}

__global__ void bn_apply_kernel(float* __restrict__ h,
                                const float* __restrict__ hn,
                                const float* __restrict__ psum, const float* __restrict__ psq,
                                const float* __restrict__ gamma, const float* __restrict__ beta,
                                int Nn)
{
    long i = (long)blockIdx.x * blockDim.x + threadIdx.x;
    if (i >= (long)Nn * 64) return;
    int c = (int)(i & 63);
    float invN = 1.0f / (float)Nn;
    float mu = psum[c] * invN;
    float var = psq[c] * invN - mu * mu;
    float rs = rsqrtf(var + 1e-5f);
    float v = (hn[i] - mu) * rs * gamma[c] + beta[c];
    h[i] = softplus_f(v) + h[i];
}

// ---------------------------------------------------------------------------
// Global mean-pool + predictor
// ---------------------------------------------------------------------------
__global__ void pool_kernel(const float* __restrict__ h, const int* __restrict__ batch,
                            float* __restrict__ sums, float* __restrict__ cnts, int Nn)
{
    long i = (long)blockIdx.x * blockDim.x + threadIdx.x;
    if (i >= (long)Nn * 64) return;
    int r = (int)(i >> 6), c = (int)(i & 63);
    int g = batch[r];
    atomicAdd(&sums[(long)g * 64 + c], h[i]);
    if (c == 0) atomicAdd(&cnts[g], 1.0f);
}

// One block per graph, 128 threads; K = input width (64 or 128).
// If cnts != nullptr, input is divided by max(cnt,1) (mean pool).
__global__ void pred_fc_kernel(const float* __restrict__ gin, const float* __restrict__ cnts,
                               const float* __restrict__ W, const float* __restrict__ b,
                               float* __restrict__ out, int K)
{
    __shared__ float gv[128];
    int g = blockIdx.x, j = threadIdx.x;
    if (j < K) {
        float v = gin[(long)g * K + j];
        if (cnts) v /= fmaxf(cnts[g], 1.0f);
        gv[j] = v;
    }
    __syncthreads();
    float acc = b[j];
    for (int k = 0; k < K; ++k) acc += gv[k] * W[k * 128 + j];
    out[(long)g * 128 + j] = softplus_f(acc);
}

__global__ void pred_out_kernel(const float* __restrict__ o2,
                                const float* __restrict__ W3, const float* __restrict__ b3,
                                float* __restrict__ out, int Gg)
{
    int g = blockIdx.x * blockDim.x + threadIdx.x;
    if (g >= Gg) return;
    float acc = b3[0];
#pragma unroll 4
    for (int k = 0; k < 128; ++k) acc += o2[(long)g * 128 + k] * W3[k];
    out[g] = acc;
}

// ---------------------------------------------------------------------------
// Host launcher
// ---------------------------------------------------------------------------
extern "C" void kernel_launch(void* const* d_in, const int* in_sizes, int n_in,
                              void* d_out, int out_size, void* d_ws, size_t ws_size,
                              hipStream_t stream)
{
    const int Nn = in_sizes[3];            // batch: [N]
    const int Ee = in_sizes[1] / 41;       // edge_attr: [E, 41]
    const int Gg = out_size;               // output: [G]
    const int Lr = 3;

    const float* x         = (const float*)d_in[0];
    const float* edge_attr = (const float*)d_in[1];
    const int*   eidx      = (const int*)d_in[2];
    const int*   rowI      = eidx;
    const int*   colI      = eidx + Ee;
    const int*   batch     = (const int*)d_in[3];
    const float* atom_W    = (const float*)d_in[4];
    const float* atom_b    = (const float*)d_in[5];
    const float* bond_W    = (const float*)d_in[6];
    const float* bond_b    = (const float*)d_in[7];
    const float* nu_W1     = (const float*)d_in[8];
    const float* nu_b1     = (const float*)d_in[9];
    const float* nu_W2     = (const float*)d_in[10];
    const float* nu_b2     = (const float*)d_in[11];
    const float* eu_W1     = (const float*)d_in[12];
    const float* eu_b1     = (const float*)d_in[13];
    const float* eu_W2     = (const float*)d_in[14];
    const float* eu_b2     = (const float*)d_in[15];
    const float* bn_gamma  = (const float*)d_in[16];
    const float* bn_beta   = (const float*)d_in[17];
    const float* pred_W1   = (const float*)d_in[18];
    const float* pred_b1   = (const float*)d_in[19];
    const float* pred_W2   = (const float*)d_in[20];
    const float* pred_b2   = (const float*)d_in[21];
    const float* pred_W3   = (const float*)d_in[22];
    const float* pred_b3   = (const float*)d_in[23];

    // --- workspace carve ---
    char* ws = (char*)d_ws;
    auto carve = [&](size_t bytes) { char* p = ws; ws += (bytes + 255) & ~(size_t)255; return p; };
    float* h     = (float*)carve((size_t)Nn * 64 * sizeof(float));
    float* hnew  = (float*)carve((size_t)Nn * 64 * sizeof(float));
    float* e     = (float*)carve((size_t)Ee * 64 * sizeof(float));
    f16*   wpack = (f16*)carve((size_t)Lr * W_TOTAL * sizeof(f16));
    float* psum  = (float*)carve(64 * sizeof(float));
    float* psq   = (float*)carve(64 * sizeof(float));
    float* sums  = (float*)carve((size_t)Gg * 64 * sizeof(float));
    float* cnts  = (float*)carve((size_t)Gg * sizeof(float));
    float* o1    = (float*)carve((size_t)Gg * 128 * sizeof(float));
    float* o2    = (float*)carve((size_t)Gg * 128 * sizeof(float));
    (void)ws_size;

    // --- weight pack (once per call) ---
    {
        int total = Lr * W_TOTAL;
        pack_weights_kernel<<<(total + 255) / 256, 256, 0, stream>>>(
            eu_W1, eu_W2, nu_W1, nu_W2, wpack, Lr);
    }

    // --- embeddings ---
    {
        long tot = (long)Nn * 64;
        atom_embed_kernel<<<(int)((tot + 255) / 256), 256, 0, stream>>>(x, atom_W, atom_b, h, Nn);
        long etot = (long)Ee * 64;
        bond_embed_kernel<<<(int)((etot + 255) / 256), 256, 0, stream>>>(edge_attr, bond_W, bond_b, e, (long)Ee);
    }

    // --- conv layers ---
    constexpr size_t SMEM_BYTES =
        (size_t)(W_TOTAL + WPB * 16 * 192 + WPB * 16 * 64) * sizeof(f16)
        + (size_t)WPB * 32 * sizeof(int);

    const int edgeBlocks = (Ee + WPB * 16 - 1) / (WPB * 16);
    const int nodeBlocks = (int)(((long)Nn * 64 + 255) / 256);

    for (int i = 0; i < Lr; ++i) {
        hipMemsetAsync(hnew, 0, (size_t)Nn * 64 * sizeof(float), stream);
        hipMemsetAsync(psum, 0, 64 * sizeof(float), stream);
        hipMemsetAsync(psq,  0, 64 * sizeof(float), stream);

        edge_layer_kernel<<<edgeBlocks, WPB * 32, SMEM_BYTES, stream>>>(
            h, e, rowI, colI,
            wpack + (size_t)i * W_TOTAL,
            eu_b1 + i * 64, eu_b2 + i * 64,
            nu_b1 + i * 64, nu_b2 + i * 64,
            hnew, Ee);

        bn_reduce_kernel<<<512, 256, 0, stream>>>(hnew, psum, psq, Nn);
        bn_apply_kernel<<<nodeBlocks, 256, 0, stream>>>(h, hnew, psum, psq,
                                                        bn_gamma + i * 64, bn_beta + i * 64, Nn);
    }

    // --- pooling + predictor ---
    hipMemsetAsync(sums, 0, (size_t)Gg * 64 * sizeof(float), stream);
    hipMemsetAsync(cnts, 0, (size_t)Gg * sizeof(float), stream);
    pool_kernel<<<nodeBlocks, 256, 0, stream>>>(h, batch, sums, cnts, Nn);

    pred_fc_kernel<<<Gg, 128, 0, stream>>>(sums, cnts, pred_W1, pred_b1, o1, 64);
    pred_fc_kernel<<<Gg, 128, 0, stream>>>(o1, nullptr, pred_W2, pred_b2, o2, 128);
    pred_out_kernel<<<(Gg + 255) / 256, 256, 0, stream>>>(o2, pred_W3, pred_b3, (float*)d_out, Gg);
}
